// QRNN_21878563406026
// MI455X (gfx1250) — matrix operand order repaired and checked by reading, not compiled, not run on hardware
//
#include <hip/hip_runtime.h>
#include <math.h>

typedef __attribute__((ext_vector_type(16))) __bf16 v16bf;
typedef __attribute__((ext_vector_type(8)))  __bf16 v8bf;
typedef __attribute__((ext_vector_type(8)))  float  v8f;
typedef __attribute__((ext_vector_type(4)))  float  v4f;
typedef __attribute__((ext_vector_type(4)))  unsigned int v4u;
typedef __attribute__((ext_vector_type(8)))  int    v8i;
typedef __attribute__((ext_vector_type(4)))  int    v4i;

#define N_MOL  256
#define N_ATOM 384
#define NN     (N_MOL * N_ATOM)   // 98304 rows
#define AEV    384
#define D1     160
#define D2     128
#define D3     96

__constant__ float c_SIG[8] = {0.5515909f, 1.8886297f, 1.3225029f, 1.2316629f,
                               2.1884933f, 1.7750372f, 1.3677907f, 1.3820058f};

__device__ __forceinline__ float celu01(float v) {
    return v > 0.f ? v : 0.1f * expm1f(v * 10.f);
}

// ---------------------------------------------------------------------------
// Stage a weight panel (nrows x kc_elems bf16 tile of a row-major [nrows x
// k_elems] matrix) into LDS at offset 0 via the Tensor Data Mover.
// D# per CDNA5 ISA ch.8: group0 = {count, lds_addr, global_addr, type=2},
// group1 = {data_size=2B, tensor_dim0=K, tensor_dim1=N, tile_dim0=KC,
// tile_dim1=N, tensor_dim0_stride=K}.  Issued by wave 0, completion via
// TENSORcnt, then workgroup barrier publishes LDS to all waves.
// ---------------------------------------------------------------------------
__device__ __forceinline__ void stage_panel(const __bf16* __restrict__ gsrc,
                                            __bf16* __restrict__ Bs,
                                            int k_elems, int kc_elems, int nrows) {
#if __has_builtin(__builtin_amdgcn_tensor_load_to_lds)
    (void)Bs;  // panel is the kernel's only __shared__ object -> LDS offset 0
    if ((threadIdx.x >> 5) == 0) {
        unsigned long long ga = (unsigned long long)gsrc;
        v4u g0;
        g0[0] = 1u;                                   // count=1 (valid D#)
        g0[1] = 0u;                                   // lds_addr (bytes)
        g0[2] = (unsigned)ga;                         // global_addr[31:0]
        g0[3] = ((unsigned)(ga >> 32) & 0x01FFFFFFu)  // global_addr[56:32]
                | (2u << 30);                         // type=2 ("image")
        v8i g1;
        g1[0] = 1 << 16;                                              // data_size=1 (2B)
        g1[1] = (k_elems & 0xFFFF) << 16;                             // tensor_dim0 lo
        g1[2] = ((k_elems >> 16) & 0xFFFF) | ((nrows & 0xFFFF) << 16);// dim0 hi | dim1 lo
        g1[3] = ((nrows >> 16) & 0xFFFF) | ((kc_elems & 0xFFFF) << 16);// dim1 hi | tile_dim0
        g1[4] = nrows & 0xFFFF;                                       // tile_dim1 (tile_dim2=0)
        g1[5] = k_elems;                                              // tensor_dim0_stride lo32
        g1[6] = 0;                                                    // stride hi | dim1_stride lo
        g1[7] = 0;
        v4i z4 = {0, 0, 0, 0};
#if defined(__clang_major__) && (__clang_major__ >= 23)
        v8i z8 = {0, 0, 0, 0, 0, 0, 0, 0};
        __builtin_amdgcn_tensor_load_to_lds(g0, g1, z4, z4, z8, 0);
#else
        __builtin_amdgcn_tensor_load_to_lds(g0, g1, z4, z4, 0);
#endif
        __builtin_amdgcn_s_wait_tensorcnt(0);
    }
    __syncthreads();
#else
    for (int c = threadIdx.x; c < nrows * kc_elems / 8; c += 256) {
        int r = c / (kc_elems / 8), kk = (c % (kc_elems / 8)) * 8;
        *(v8bf*)(Bs + r * kc_elems + kk) =
            *(const v8bf*)(gsrc + (size_t)r * k_elems + kk);
    }
    __syncthreads();
#endif
}

// ---------------------------------------------------------------------------
// Weight prep: f32 -> bf16 copies, split W1 into AEV part + (q, esp) columns
// ---------------------------------------------------------------------------
__global__ void prep_weights(const float* __restrict__ W1,
                             const float* __restrict__ W2,
                             const float* __restrict__ W3,
                             __bf16* __restrict__ W1a,
                             float* __restrict__ w1q, float* __restrict__ w1e,
                             __bf16* __restrict__ W2b, __bf16* __restrict__ W3b) {
    int idx = blockIdx.x * 256 + threadIdx.x;
    if (idx < D1 * AEV) {
        int n = idx / AEV, k = idx % AEV;
        W1a[idx] = (__bf16)W1[n * (AEV + 2) + k];
    }
    int i2 = idx - D1 * AEV;
    if (i2 >= 0 && i2 < D1) {
        w1q[i2] = W1[i2 * (AEV + 2) + AEV];
        w1e[i2] = W1[i2 * (AEV + 2) + AEV + 1];
    }
    int i3 = idx - (D1 * AEV + D1);
    if (i3 >= 0 && i3 < D2 * D1) W2b[i3] = (__bf16)W2[i3];
    int i4 = idx - (D1 * AEV + D1 + D2 * D1);
    if (i4 >= 0 && i4 < D3 * D2) W3b[i4] = (__bf16)W3[i4];
}

__global__ void init_zero(float* __restrict__ charges, float* __restrict__ esp) {
    int i = blockIdx.x * 256 + threadIdx.x;
    if (i < NN) { charges[i] = 0.f; esp[i] = 0.f; }
}

// ---------------------------------------------------------------------------
// Layer 1 (once per call): z1 = aev @ W1aᵀ + b1.
// Each wave owns 16 rows and ALL 10 N-tiles (acc in 80 VGPRs) so the f32 AEV
// stream is read exactly once.  Weights staged K-chunk-wise via TDM.
// grid = NN/128, block = 256 (8 waves)
// ---------------------------------------------------------------------------
__global__ __launch_bounds__(256) void gemm1_kernel(
        const float* __restrict__ aev, const __bf16* __restrict__ W1a,
        const float* __restrict__ b1, float* __restrict__ z1) {
    constexpr int K = AEV, N = D1, KC = 128, NT = N / 16;
    __shared__ __bf16 Bs[KC * N];   // [n][kc] : n*KC + kk
    int tid = threadIdx.x;
    int wave = tid >> 5, lane = tid & 31, half = lane >> 4, l = lane & 15;
    int m0 = blockIdx.x * 128 + wave * 16;
    const float* arow = aev + (size_t)(m0 + l) * K;

    const v8f vzero = {0.f, 0.f, 0.f, 0.f, 0.f, 0.f, 0.f, 0.f};
    v8f acc[NT];
#pragma unroll
    for (int t = 0; t < NT; ++t) acc[t] = vzero;

    for (int kc = 0; kc < K; kc += KC) {
        stage_panel(W1a + kc, Bs, K, KC, N);
#pragma unroll
        for (int k0 = 0; k0 < KC; k0 += 32) {
            union { v16bf v; __bf16 e[16]; } a;
            int kg = kc + k0;
            // A layout: lanes 0-15 hold K = +0..7 and +16..23; lanes 16-31: +8
            v4f p0 = *(const v4f*)(arow + kg + half * 8);
            v4f p1 = *(const v4f*)(arow + kg + half * 8 + 4);
            v4f p2 = *(const v4f*)(arow + kg + 16 + half * 8);
            v4f p3 = *(const v4f*)(arow + kg + 16 + half * 8 + 4);
#pragma unroll
            for (int i = 0; i < 4; ++i) {
                a.e[i]      = (__bf16)p0[i];
                a.e[4 + i]  = (__bf16)p1[i];
                a.e[8 + i]  = (__bf16)p2[i];
                a.e[12 + i] = (__bf16)p3[i];
            }
#pragma unroll
            for (int nt = 0; nt < NT; ++nt) {
                const __bf16* brow = Bs + (nt * 16 + l) * KC;
                union { v16bf v; v8bf h[2]; } b;
                // B layout: K = k0 + half*8 + 0..7 and k0 + 16 + half*8 + 0..7
                b.h[0] = *(const v8bf*)(brow + k0 + half * 8);
                b.h[1] = *(const v8bf*)(brow + k0 + 16 + half * 8);
                acc[nt] = __builtin_amdgcn_wmma_f32_16x16x32_bf16(
                        false, a.v, false, b.v, (short)0, acc[nt], false, false);
            }
        }
        __syncthreads();
    }
#pragma unroll
    for (int nt = 0; nt < NT; ++nt) {
        int col = nt * 16 + l;
        float bc = b1[col];
#pragma unroll
        for (int r = 0; r < 8; ++r)
            z1[(size_t)(m0 + half * 8 + r) * N + col] = acc[nt][r] + bc;
    }
}

// ---------------------------------------------------------------------------
// Layers 2/3: h_out = celu(h_in @ Wᵀ + b).  Full weight matrix staged once in
// LDS by TDM; each wave holds all N-tiles so A is read exactly once.
// ---------------------------------------------------------------------------
template <int K, int N>
__global__ __launch_bounds__(256) void gemm_bf16_celu(
        const __bf16* __restrict__ A, const __bf16* __restrict__ W,
        const float* __restrict__ bias, __bf16* __restrict__ out) {
    constexpr int NT = N / 16;
    __shared__ __bf16 Bs[K * N];    // [n][k] : n*K + k
    int tid = threadIdx.x;
    int wave = tid >> 5, lane = tid & 31, half = lane >> 4, l = lane & 15;
    int m0 = blockIdx.x * 128 + wave * 16;
    const __bf16* arow = A + (size_t)(m0 + l) * K;

    stage_panel(W, Bs, K, K, N);

    const v8f vzero = {0.f, 0.f, 0.f, 0.f, 0.f, 0.f, 0.f, 0.f};
    v8f acc[NT];
#pragma unroll
    for (int t = 0; t < NT; ++t) acc[t] = vzero;

#pragma unroll
    for (int k0 = 0; k0 < K; k0 += 32) {
        union { v16bf v; v8bf h[2]; } a;
        a.h[0] = *(const v8bf*)(arow + k0 + half * 8);
        a.h[1] = *(const v8bf*)(arow + k0 + 16 + half * 8);
#pragma unroll
        for (int nt = 0; nt < NT; ++nt) {
            const __bf16* brow = Bs + (nt * 16 + l) * K;
            union { v16bf v; v8bf h[2]; } b;
            b.h[0] = *(const v8bf*)(brow + k0 + half * 8);
            b.h[1] = *(const v8bf*)(brow + k0 + 16 + half * 8);
            acc[nt] = __builtin_amdgcn_wmma_f32_16x16x32_bf16(
                    false, a.v, false, b.v, (short)0, acc[nt], false, false);
        }
    }
#pragma unroll
    for (int nt = 0; nt < NT; ++nt) {
        int col = nt * 16 + l;
        float bc = bias[col];
#pragma unroll
        for (int r = 0; r < 8; ++r)
            out[(size_t)(m0 + half * 8 + r) * N + col] =
                (__bf16)celu01(acc[nt][r] + bc);
    }
}

// ---------------------------------------------------------------------------
// Per-iteration rank-2 layer-1 update: h1 = celu(z1_base + q*w1q + esp*w1e)
// ---------------------------------------------------------------------------
__global__ void h1_update(const float* __restrict__ z1,
                          const float* __restrict__ charges,
                          const float* __restrict__ esp,
                          const float* __restrict__ w1q,
                          const float* __restrict__ w1e,
                          __bf16* __restrict__ h1) {
    size_t idx = (size_t)blockIdx.x * 256 + threadIdx.x;
    if (idx >= (size_t)NN * D1) return;
    int row = (int)(idx / D1), col = (int)(idx % D1);
    float v = z1[idx] + charges[row] * w1q[col] + esp[row] * w1e[col];
    h1[idx] = (__bf16)celu01(v);
}

// ---------------------------------------------------------------------------
// Layer 4: chi = softplus(h3 @ W4ᵀ + b4), masked by species
// ---------------------------------------------------------------------------
__global__ void chi_kernel(const __bf16* __restrict__ h3,
                           const float* __restrict__ W4, const float* __restrict__ b4,
                           const int* __restrict__ species, float* __restrict__ chi) {
    int row = blockIdx.x * 256 + threadIdx.x;
    if (row >= NN) return;
    const __bf16* hp = h3 + (size_t)row * D3;
    float acc = b4[0];
#pragma unroll 8
    for (int k = 0; k < D3; ++k) acc += (float)hp[k] * W4[k];
    float sp = acc > 20.f ? acc : log1pf(expf(acc));
    chi[row] = (species[row] != -1) ? sp : 0.f;
}

// ---------------------------------------------------------------------------
// Per-molecule reduction + charge update. One block (384 thr) per molecule.
// ---------------------------------------------------------------------------
__global__ __launch_bounds__(384) void mol_kernel(
        const float* __restrict__ chi, const int* __restrict__ species,
        const float* __restrict__ net_charge, float* __restrict__ charges) {
    __shared__ float s_chi[N_ATOM];
    __shared__ float s_cnt[N_ATOM];
    int mol = blockIdx.x, a = threadIdx.x;
    int base = mol * N_ATOM;
    float mask = (species[base + a] != -1) ? 1.f : 0.f;
    float my_chi = chi[base + a];
    s_chi[a] = my_chi;
    s_cnt[a] = mask;
    __syncthreads();
    for (int s = 192; s >= 3; s >>= 1) {
        if (a < s) { s_chi[a] += s_chi[a + s]; s_cnt[a] += s_cnt[a + s]; }
        __syncthreads();
    }
    float chi_sum = s_chi[0] + s_chi[1] + s_chi[2];
    float cnt     = s_cnt[0] + s_cnt[1] + s_cnt[2];
    float qn       = net_charge[mol];
    float k_net    = 1.f + fabsf(qn) / chi_sum;
    float chi_mean = chi_sum / cnt;
    float k_p = (qn > 0.f) ? k_net : 1.f;
    float k_n = (qn < 0.f) ? k_net : 1.f;
    charges[base + a] = (-k_n * my_chi + k_p * chi_mean) * mask;
}

// ---------------------------------------------------------------------------
// ESP: esp_j = sum_i q_i * erf(d_ij/sqrt(2(s_i²+s_j²)))/d_ij (i != j).
// One block (384 thr) per molecule; j_ij recomputed on the fly (erf VALU).
// Padded i atoms contribute 0 automatically (q_i == 0).
// ---------------------------------------------------------------------------
__global__ __launch_bounds__(384) void esp_kernel(
        const float* __restrict__ coords, const int* __restrict__ species,
        const float* __restrict__ charges, float* __restrict__ esp) {
    __shared__ float cx[N_ATOM], cy[N_ATOM], cz[N_ATOM], sg[N_ATOM], sq[N_ATOM];
    int mol = blockIdx.x, j = threadIdx.x;
    int base = mol * N_ATOM;
    int sp = species[base + j];
    const float* cp = coords + (size_t)(base + j) * 3;
    cx[j] = cp[0]; cy[j] = cp[1]; cz[j] = cp[2];
    sg[j] = c_SIG[sp < 0 ? 7 : sp];
    sq[j] = charges[base + j];
    __syncthreads();
    if (sp == -1) { esp[base + j] = 0.f; return; }
    float xj = cx[j], yj = cy[j], zj = cz[j];
    float s2j = sg[j] * sg[j];
    const float invA0 = 1.f / 0.529177249f;
    float acc = 0.f;
#pragma unroll 4
    for (int i = 0; i < N_ATOM; ++i) {
        float dx = cx[i] - xj, dy = cy[i] - yj, dz = cz[i] - zj;
        float d  = sqrtf(dx * dx + dy * dy + dz * dz + 1e-16f) * invA0;
        float s2 = fmaxf(sg[i] * sg[i] + s2j, 1e-8f);
        float t  = erff(d * rsqrtf(2.f * s2)) / d;
        acc += (i != j) ? sq[i] * t : 0.f;
    }
    esp[base + j] = acc;
}

// ---------------------------------------------------------------------------
extern "C" void kernel_launch(void* const* d_in, const int* in_sizes, int n_in,
                              void* d_out, int out_size, void* d_ws, size_t ws_size,
                              hipStream_t stream) {
    const int*   species = (const int*)  d_in[0];
    const float* coords  = (const float*)d_in[1];
    const float* netq    = (const float*)d_in[2];
    const float* aev     = (const float*)d_in[3];
    const float* W1 = (const float*)d_in[4];
    const float* b1 = (const float*)d_in[5];
    const float* W2 = (const float*)d_in[6];
    const float* b2 = (const float*)d_in[7];
    const float* W3 = (const float*)d_in[8];
    const float* b3 = (const float*)d_in[9];
    const float* W4 = (const float*)d_in[10];
    const float* b4 = (const float*)d_in[11];

    char* w = (char*)d_ws;
    float*  z1      = (float*) w; w += (size_t)NN * D1 * 4;
    __bf16* h1      = (__bf16*)w; w += (size_t)NN * D1 * 2;
    __bf16* h2      = (__bf16*)w; w += (size_t)NN * D2 * 2;
    __bf16* h3      = (__bf16*)w; w += (size_t)NN * D3 * 2;
    float*  chi     = (float*) w; w += (size_t)NN * 4;
    float*  charges = (float*) w; w += (size_t)NN * 4;
    float*  esp     = (float*) w; w += (size_t)NN * 4;
    __bf16* W1a     = (__bf16*)w; w += (size_t)D1 * AEV * 2;
    float*  w1q     = (float*) w; w += (size_t)D1 * 4;
    float*  w1e     = (float*) w; w += (size_t)D1 * 4;
    __bf16* W2b     = (__bf16*)w; w += (size_t)D2 * D1 * 2;
    __bf16* W3b     = (__bf16*)w; w += (size_t)D3 * D2 * 2;

    int prep_n = D1 * AEV + D1 + D2 * D1 + D3 * D2;
    prep_weights<<<(prep_n + 255) / 256, 256, 0, stream>>>(W1, W2, W3, W1a, w1q, w1e, W2b, W3b);
    init_zero<<<(NN + 255) / 256, 256, 0, stream>>>(charges, esp);

    // One-time layer-1 base GEMM (iteration-invariant AEV contribution)
    gemm1_kernel<<<NN / 128, 256, 0, stream>>>(aev, W1a, b1, z1);

    for (int it = 0; it < 4; ++it) {
        h1_update<<<(int)(((size_t)NN * D1 + 255) / 256), 256, 0, stream>>>(
            z1, charges, esp, w1q, w1e, h1);
        gemm_bf16_celu<D1, D2><<<NN / 128, 256, 0, stream>>>(h1, W2b, b2, h2);
        gemm_bf16_celu<D2, D3><<<NN / 128, 256, 0, stream>>>(h2, W3b, b3, h3);
        chi_kernel<<<(NN + 255) / 256, 256, 0, stream>>>(h3, W4, b4, species, chi);
        mol_kernel<<<N_MOL, N_ATOM, 0, stream>>>(chi, species, netq, charges);
        if (it < 3)  // ESP of the final iteration is dead code
            esp_kernel<<<N_MOL, N_ATOM, 0, stream>>>(coords, species, charges, esp);
    }

    // Output tuple: (species, pred_charges) concatenated flat
    hipMemcpyAsync(d_out, d_in[0], (size_t)NN * 4, hipMemcpyDeviceToDevice, stream);
    hipMemcpyAsync((char*)d_out + (size_t)NN * 4, charges, (size_t)NN * 4,
                   hipMemcpyDeviceToDevice, stream);
}
